// TGN_18537078849943
// MI455X (gfx1250) — matrix operand order repaired and checked by reading, not compiled, not run on hardware
//
#include <hip/hip_runtime.h>

// ---------------------------------------------------------------------------
// MI455X (gfx1250) TGN inference.
//   * bf16 activations/weights, f32 accumulation via v_wmma_f32_16x16x32_bf16
//   * algebraic re-association removes the per-neighbor Wk/Wv GEMMs:
//       score = (Wk_h^T qh) . kmat         (p-GEMM,   M=n rows per head)
//       out_h = (sum_k a_k kmat_k) @ Wv_h  (mix-GEMM, M=n rows per head)
//     -> ~77 GFLOP of GEMM vs ~420 naive; ~0.8 GB of gathers at 23.3 TB/s.
//   * all weights stored B^T ([N][K], bf16) so both WMMA fragments load as
//     two GLOBAL_LOAD_B128 per lane (no predication, no scalar u16 loads).
// ---------------------------------------------------------------------------

typedef __attribute__((ext_vector_type(16))) __bf16 v16bf;
typedef __attribute__((ext_vector_type(8)))  __bf16 v8bf;
typedef __attribute__((ext_vector_type(8)))  float  v8f;

#define GF_RELU    1
#define GF_OUTBF16 2

#define NB   2048     // rows per pipeline invocation (call 2 runs 10 chunks)
#define KNB  10
#define FDIM 256
#define DQv  512
#define DKv  768

// ---------------- elementwise / gather kernels -----------------------------

__global__ void __launch_bounds__(256) k_cvt_bf16(const float* __restrict__ s,
                                                  __bf16* __restrict__ d, int n) {
  int i = blockIdx.x * blockDim.x + threadIdx.x;
  if (i < n) d[i] = (__bf16)s[i];
}

// transpose-convert: dst[N][K] (bf16) = src[K][N] (f32)
__global__ void __launch_bounds__(256) k_cvt_t(const float* __restrict__ s,
                                               __bf16* __restrict__ d, int K, int N) {
  int gid = blockIdx.x * blockDim.x + threadIdx.x;
  if (gid >= K * N) return;
  int n = gid / K, k = gid % K;
  d[gid] = (__bf16)s[(size_t)k * N + n];
}

// q[i, 0:256] = src_feat (gather feat(idx)=memory+node_feat, or direct f32)
// q[i, 256:512] = cos(time_b[j])   (tenc of t=0)
__global__ void __launch_bounds__(256) k_build_q(const int* __restrict__ srcIdx,
                                                 const float* __restrict__ srcDirect,
                                                 const float* __restrict__ mem,
                                                 const float* __restrict__ nodef,
                                                 const float* __restrict__ time_b,
                                                 __bf16* __restrict__ q, int n) {
  int gid = blockIdx.x * blockDim.x + threadIdx.x;
  if (gid >= n * DQv) return;
  int i = gid >> 9, j = gid & 511;
  float v;
  if (j < FDIM) {
    if (srcIdx) { size_t b = (size_t)srcIdx[i] * FDIM + j; v = mem[b] + nodef[b]; }
    else        { v = srcDirect[(size_t)i * FDIM + j]; }
  } else {
    v = cosf(time_b[j - FDIM]);
  }
  q[gid] = (__bf16)v;
}

// kmat[i,k, :] = [ nbr_feat(256) | edge_feat(256) | cos((srcT-eT)*w + b)(256) ]
__global__ void __launch_bounds__(256) k_build_kmat(const int* __restrict__ nbrIdx,
                                                    const float* __restrict__ nbrDirect,
                                                    const int* __restrict__ eIdx,
                                                    const float* __restrict__ eT,
                                                    const float* __restrict__ srcT,
                                                    const float* __restrict__ mem,
                                                    const float* __restrict__ nodef,
                                                    const float* __restrict__ edgef,
                                                    const float* __restrict__ time_w,
                                                    const float* __restrict__ time_b,
                                                    __bf16* __restrict__ kmat, int n) {
  int gid = blockIdx.x * blockDim.x + threadIdx.x;
  if (gid >= n * KNB * DKv) return;
  int j   = gid % DKv;
  int row = gid / DKv;
  int i   = row / KNB;
  float v;
  if (j < FDIM) {
    if (nbrDirect) v = nbrDirect[(size_t)row * FDIM + j];
    else { size_t b = (size_t)nbrIdx[row] * FDIM + j; v = mem[b] + nodef[b]; }
  } else if (j < 2 * FDIM) {
    v = edgef[(size_t)eIdx[row] * FDIM + (j - FDIM)];
  } else {
    int jj = j - 2 * FDIM;
    float dt = srcT[i] - eT[row];
    v = cosf(dt * time_w[jj] + time_b[jj]);
  }
  kmat[gid] = (__bf16)v;
}

// ---------------- generic bf16 WMMA GEMM -----------------------------------
// C[M,N] = act( A[M,K]bf16 @ B[K,N]bf16 + bias[N] ), B given TRANSPOSED [N][K].
// one wave -> 16x64 tile (4 accumulators), K stepped by 32.
// Preconditions (guaranteed by caller): M%16==0, N%64==0, K%32==0,
// A/BT rows 16B-aligned -> all fragment loads are unpredicated B128 loads.
// Fragment layouts per CDNA5 ISA 7.12.2 (wave32):
//   A 16x32: lane&15 = row M; half-lane picks K {0..7,16..23} vs {8..15,24..31}
//   B 32x16: lane&15 = col N; half-lane picks K {0..7,16..23} vs {8..15,24..31}
//   C 16x16: lane&15 = N; (lane>>4)*8 + vgpr = M
__global__ void __launch_bounds__(256) k_gemm_bf16(const __bf16* __restrict__ A, int lda,
                                                   const __bf16* __restrict__ BT, int ldb,
                                                   void* __restrict__ Cout, int ldc,
                                                   const float* __restrict__ bias,
                                                   int M, int N, int K, int flags) {
  const int lane   = threadIdx.x & 31;
  const int tilesN = N >> 6;
  int tile = blockIdx.x * 8 + (threadIdx.x >> 5);
  if (tile >= (M >> 4) * tilesN) return;   // wave-uniform: EXEC stays all-ones
  const int tm = tile / tilesN, tn = tile % tilesN;

  const int aKh = (lane >> 4) << 3;        // 0 or 8
  const int bKo = (lane >> 4) << 3;        // 0 or 8
  const __bf16* Arow = A + (size_t)(tm * 16 + (lane & 15)) * lda + aKh;
  const __bf16* Brow = BT + (size_t)(tn * 64 + (lane & 15)) * ldb + bKo;

  v8f acc0 = {}, acc1 = {}, acc2 = {}, acc3 = {};
  for (int k0 = 0; k0 < K; k0 += 32) {
    v8bf alo = *(const v8bf*)(Arow + k0);
    v8bf ahi = *(const v8bf*)(Arow + k0 + 16);
    v16bf af = __builtin_shufflevector(alo, ahi, 0, 1, 2, 3, 4, 5, 6, 7,
                                       8, 9, 10, 11, 12, 13, 14, 15);
#define BSTEP(ACC, T)                                                          \
    {                                                                          \
      const __bf16* Bp = Brow + (size_t)(T) * 16 * ldb + k0;                   \
      v8bf blo = *(const v8bf*)(Bp);                                           \
      v8bf bhi = *(const v8bf*)(Bp + 16);                                      \
      v16bf bfr = __builtin_shufflevector(blo, bhi, 0, 1, 2, 3, 4, 5, 6, 7,    \
                                          8, 9, 10, 11, 12, 13, 14, 15);       \
      ACC = __builtin_amdgcn_wmma_f32_16x16x32_bf16(false, af, false, bfr,     \
                                                    (short)0, ACC, false, false); \
    }
    BSTEP(acc0, 0) BSTEP(acc1, 1) BSTEP(acc2, 2) BSTEP(acc3, 3)
#undef BSTEP
  }

  const int cn_l = lane & 15;
  const int cm0  = tm * 16 + ((lane >> 4) << 3);
#define CSTORE(ACC, T)                                                         \
  {                                                                            \
    int cn = tn * 64 + (T) * 16 + cn_l;                                        \
    float bb = bias ? bias[cn] : 0.0f;                                         \
    _Pragma("unroll")                                                          \
    for (int r = 0; r < 8; r++) {                                              \
      float v = ACC[r] + bb;                                                   \
      if (flags & GF_RELU) v = fmaxf(v, 0.0f);                                 \
      if (flags & GF_OUTBF16) ((__bf16*)Cout)[(size_t)(cm0 + r) * ldc + cn] = (__bf16)v; \
      else                    ((float*)Cout)[(size_t)(cm0 + r) * ldc + cn]  = v;         \
    }                                                                          \
  }
  CSTORE(acc0, 0) CSTORE(acc1, 1) CSTORE(acc2, 2) CSTORE(acc3, 3)
#undef CSTORE
}

// ---------------- attention: scores -> softmax -> kmat mixing --------------
// one wave per (src i, head h); lanes split the 768-dim axis, butterfly-reduce.
__global__ void __launch_bounds__(256) k_attn(const __bf16* __restrict__ kmat,
                                              const float* __restrict__ p,
                                              const __bf16* __restrict__ qh,
                                              const float* __restrict__ bk,
                                              const int* __restrict__ nbrIdx,
                                              __bf16* __restrict__ mmix,
                                              int* __restrict__ amsk, int n) {
  int wid  = (blockIdx.x * blockDim.x + threadIdx.x) >> 5;
  int lane = threadIdx.x & 31;
  if (wid >= n * 2) return;
  int i = wid >> 1, h = wid & 1;
  const __bf16* kb = kmat + (size_t)i * KNB * DKv;
  const float*  pr = p + (size_t)i * 2 * DKv + h * DKv;
  float pv[24];
#pragma unroll
  for (int t = 0; t < 24; t++) pv[t] = pr[lane + 32 * t];

  // bk contribution: qh_h . bk_h (bk is zero in setup but kept for fidelity)
  const __bf16* qr = qh + (size_t)i * DQv + h * FDIM;
  const float* bkh = bk + h * FDIM;
  float bkd = 0.f;
#pragma unroll
  for (int t = 0; t < 8; t++) { int d = lane + 32 * t; bkd += (float)qr[d] * bkh[d]; }
#pragma unroll
  for (int o = 16; o; o >>= 1) bkd += __shfl_xor(bkd, o, 32);

  float s[KNB];
  int zc = 0;
#pragma unroll
  for (int k = 0; k < KNB; k++) {
    const __bf16* kr = kb + (size_t)k * DKv;
    float acc = 0.f;
#pragma unroll
    for (int t = 0; t < 24; t++) acc += pv[t] * (float)kr[lane + 32 * t];
#pragma unroll
    for (int o = 16; o; o >>= 1) acc += __shfl_xor(acc, o, 32);
    float sc = (acc + bkd) * 0.0625f;                // 1/sqrt(dh=256)
    if (nbrIdx[(size_t)i * KNB + k] == 0) { sc = -1e9f; zc++; }
    s[k] = sc;
  }
  float mx = s[0];
#pragma unroll
  for (int k = 1; k < KNB; k++) mx = fmaxf(mx, s[k]);
  float e[KNB], den = 0.f;
#pragma unroll
  for (int k = 0; k < KNB; k++) { e[k] = expf(s[k] - mx); den += e[k]; }
  float inv = 1.f / den;

  __bf16* mr = mmix + (size_t)wid * DKv;             // [n, H, 768]
#pragma unroll
  for (int t = 0; t < 24; t++) {
    int d = lane + 32 * t;
    float a = 0.f;
#pragma unroll
    for (int k = 0; k < KNB; k++) a += e[k] * (float)kb[(size_t)k * DKv + d];
    mr[d] = (__bf16)(a * inv);
  }
  if (h == 0 && lane == 0) amsk[i] = (zc == KNB) ? 1 : 0;
}

// zero fully-masked attn rows (cols 0..511) + append src_feat (cols 512..767)
__global__ void __launch_bounds__(256) k_finish_h1in(__bf16* __restrict__ h1in,
                                                     const __bf16* __restrict__ q,
                                                     const int* __restrict__ amsk, int n) {
  int gid = blockIdx.x * blockDim.x + threadIdx.x;
  if (gid >= n * DKv) return;
  int i = gid / DKv, j = gid % DKv;
  if (j >= DQv) h1in[gid] = q[(size_t)i * DQv + (j - DQv)];
  else if (amsk[i]) h1in[gid] = (__bf16)0.0f;
}

// projector tail: logits = hp @ Wp2 + bp2 (256x5), row softmax
__global__ void __launch_bounds__(256) k_final(const float* __restrict__ hp,
                                               const float* __restrict__ Wp2,
                                               const float* __restrict__ bp2,
                                               float* __restrict__ out, int n) {
  int wid  = (blockIdx.x * blockDim.x + threadIdx.x) >> 5;
  int lane = threadIdx.x & 31;
  if (wid >= n) return;
  const float* hr = hp + (size_t)wid * FDIM;
  float lg[5];
#pragma unroll
  for (int c = 0; c < 5; c++) {
    float acc = 0.f;
#pragma unroll
    for (int t = 0; t < 8; t++) { int d = lane + 32 * t; acc += hr[d] * Wp2[d * 5 + c]; }
#pragma unroll
    for (int o = 16; o; o >>= 1) acc += __shfl_xor(acc, o, 32);
    lg[c] = acc + bp2[c];
  }
  float mx = lg[0];
  for (int c = 1; c < 5; c++) mx = fmaxf(mx, lg[c]);
  float den = 0.f, mine = 0.f;
  for (int c = 0; c < 5; c++) { float ec = expf(lg[c] - mx); den += ec; if (lane == c) mine = ec; }
  if (lane < 5) out[(size_t)wid * 5 + lane] = mine / den;
}

// ---------------- host orchestration ---------------------------------------

static inline void launch_gemm(hipStream_t st, const __bf16* A, int lda,
                               const __bf16* BT, int ldb, void* C, int ldc,
                               const float* bias, int M, int N, int K, int flags) {
  int blocks = ((M >> 4) * (N >> 6) + 7) >> 3;
  k_gemm_bf16<<<blocks, 256, 0, st>>>(A, lda, BT, ldb, C, ldc, bias, M, N, K, flags);
}

extern "C" void kernel_launch(void* const* d_in, const int* in_sizes, int n_in,
                              void* d_out, int out_size, void* d_ws, size_t ws_size,
                              hipStream_t stream) {
  (void)in_sizes; (void)n_in; (void)out_size; (void)ws_size;
  const int*   nodes         = (const int*)  d_in[0];
  const float* timestamps    = (const float*)d_in[1];
  const int*   neighbors1    = (const int*)  d_in[2];
  const int*   edge_idxs1    = (const int*)  d_in[3];
  const float* edge_times1   = (const float*)d_in[4];
  const int*   neighbors2    = (const int*)  d_in[5];
  const int*   edge_idxs2    = (const int*)  d_in[6];
  const float* edge_times2   = (const float*)d_in[7];
  const float* node_features = (const float*)d_in[8];
  const float* edge_features = (const float*)d_in[9];
  const float* memory        = (const float*)d_in[10];
  const float* time_w        = (const float*)d_in[11];
  const float* time_b        = (const float*)d_in[12];
  const float* Wq  = (const float*)d_in[13]; const float* bq  = (const float*)d_in[14];
  const float* Wk  = (const float*)d_in[15]; const float* bk  = (const float*)d_in[16];
  const float* Wv  = (const float*)d_in[17]; const float* bv  = (const float*)d_in[18];
  const float* Wo  = (const float*)d_in[19]; const float* bo  = (const float*)d_in[20];
  const float* W1  = (const float*)d_in[21]; const float* b1  = (const float*)d_in[22];
  const float* W2  = (const float*)d_in[23]; const float* b2  = (const float*)d_in[24];
  const float* Wp1 = (const float*)d_in[25]; const float* bp1 = (const float*)d_in[26];
  const float* Wp2 = (const float*)d_in[27]; const float* bp2 = (const float*)d_in[28];

  // bump allocator over d_ws (total ~97 MB)
  char* wp = (char*)d_ws;
  auto alloc = [&](size_t b) -> char* { char* r = wp; wp += (b + 255) & ~(size_t)255; return r; };
  __bf16* WqT  = (__bf16*)alloc((size_t)2 * DQv * DQv * 2);   // [l][N=512][K=512]
  __bf16* WkB  = (__bf16*)alloc((size_t)2 * DKv * DQv * 2);   // [l][768][512] (as-is == BT of Wk_h^T)
  __bf16* WvT  = (__bf16*)alloc((size_t)2 * DQv * DKv * 2);   // [l][N=512][K=768]
  __bf16* WoT  = (__bf16*)alloc((size_t)2 * DQv * DQv * 2);   // [l][512][512]
  __bf16* W1T  = (__bf16*)alloc((size_t)2 * FDIM * DKv * 2);  // [l][256][768]
  __bf16* W2T  = (__bf16*)alloc((size_t)2 * FDIM * FDIM * 2); // [l][256][256]
  __bf16* Wp1T = (__bf16*)alloc((size_t)FDIM * FDIM * 2);     // [256][256]
  __bf16* q    = (__bf16*)alloc((size_t)NB * DQv * 2);
  __bf16* qh   = (__bf16*)alloc((size_t)NB * DQv * 2);
  float*  p    = (float*) alloc((size_t)NB * 2 * DKv * 4);
  __bf16* kmat = (__bf16*)alloc((size_t)NB * KNB * DKv * 2);
  __bf16* mmix = (__bf16*)alloc((size_t)NB * 2 * DKv * 2);
  __bf16* vout = (__bf16*)alloc((size_t)NB * DQv * 2);
  __bf16* h1in = (__bf16*)alloc((size_t)NB * DKv * 2);
  __bf16* hbuf = (__bf16*)alloc((size_t)NB * FDIM * 2);
  int*    amsk = (int*)   alloc((size_t)NB * 4);
  float*  src1 = (float*) alloc((size_t)NB * FDIM * 4);
  float*  nbr1 = (float*) alloc((size_t)NB * KNB * FDIM * 4);
  float*  emb  = (float*) alloc((size_t)NB * FDIM * 4);
  __bf16* embB = (__bf16*)alloc((size_t)NB * FDIM * 2);
  float*  hp   = (float*) alloc((size_t)NB * FDIM * 4);

  // weights -> bf16 (transposed to [N][K] for the WMMA B-path)
  auto cvtT = [&](const float* s, __bf16* d, int K, int N) {
    k_cvt_t<<<(K * N + 255) / 256, 256, 0, stream>>>(s, d, K, N);
  };
  for (int l = 0; l < 2; l++) {
    cvtT(Wq + (size_t)l * DQv * DQv,  WqT + (size_t)l * DQv * DQv,  DQv, DQv);
    cvtT(Wv + (size_t)l * DKv * DQv,  WvT + (size_t)l * DQv * DKv,  DKv, DQv);
    cvtT(Wo + (size_t)l * DQv * DQv,  WoT + (size_t)l * DQv * DQv,  DQv, DQv);
    cvtT(W1 + (size_t)l * DKv * FDIM, W1T + (size_t)l * FDIM * DKv, DKv, FDIM);
    cvtT(W2 + (size_t)l * FDIM * FDIM, W2T + (size_t)l * FDIM * FDIM, FDIM, FDIM);
  }
  cvtT(Wp1, Wp1T, FDIM, FDIM);
  k_cvt_bf16<<<(2 * DKv * DQv + 255) / 256, 256, 0, stream>>>(Wk, WkB, 2 * DKv * DQv);

  // one temporal-attention pipeline over NB=2048 source rows
  auto run_attn = [&](int l, const int* srcIdx, const float* srcDirect, const float* srcT,
                      const int* nbrIdx, const float* nbrDirect,
                      const int* eIdx, const float* eT, float* dest) {
    const int n = NB;
    k_build_q<<<(n * DQv + 255) / 256, 256, 0, stream>>>(
        srcIdx, srcDirect, memory, node_features, time_b, q, n);
    // qh = q @ Wq + bq  (bf16 out: feeds p-GEMM as A)
    launch_gemm(stream, q, DQv, WqT + (size_t)l * DQv * DQv, DQv,
                qh, DQv, bq + l * DQv, n, DQv, DQv, GF_OUTBF16);
    // p[:,h,:] = qh_h @ Wk_h^T ; BT of (Wk_h^T) is Wk itself: rows c, cols d
    for (int h = 0; h < 2; h++)
      launch_gemm(stream, qh + h * FDIM, DQv, WkB + (size_t)l * DKv * DQv + h * FDIM, DQv,
                  p + h * DKv, 2 * DKv, nullptr, n, DKv, FDIM, 0);
    k_build_kmat<<<(n * KNB * DKv + 255) / 256, 256, 0, stream>>>(
        nbrIdx, nbrDirect, eIdx, eT, srcT, memory, node_features, edge_features,
        time_w, time_b, kmat, n);
    k_attn<<<(2 * n * 32 + 255) / 256, 256, 0, stream>>>(
        kmat, p, qh, bk + l * DQv, nbrIdx, mmix, amsk, n);
    // out_h = mmix_h @ Wv_h + bv_h  -> vout[:, h*256:...]
    for (int h = 0; h < 2; h++)
      launch_gemm(stream, mmix + h * DKv, 2 * DKv,
                  WvT + ((size_t)l * DQv + h * FDIM) * DKv, DKv,
                  vout + h * FDIM, DQv, bv + l * DQv + h * FDIM, n, FDIM, DKv, GF_OUTBF16);
    // (attn out) @ Wo + bo -> h1in cols 0..511 (ldc=768)
    launch_gemm(stream, vout, DQv, WoT + (size_t)l * DQv * DQv, DQv,
                h1in, DKv, bo + l * DQv, n, DQv, DQv, GF_OUTBF16);
    k_finish_h1in<<<(n * DKv + 255) / 256, 256, 0, stream>>>(h1in, q, amsk, n);
    launch_gemm(stream, h1in, DKv, W1T + (size_t)l * FDIM * DKv, DKv,
                hbuf, FDIM, b1 + l * FDIM, n, FDIM, DKv, GF_RELU | GF_OUTBF16);
    launch_gemm(stream, hbuf, FDIM, W2T + (size_t)l * FDIM * FDIM, FDIM,
                dest, FDIM, b2 + l * FDIM, n, FDIM, FDIM, 0);
  };

  // call 1: layer-0 embedding of the batch nodes
  run_attn(0, nodes, nullptr, timestamps, neighbors1, nullptr,
           edge_idxs1, edge_times1, src1);
  // call 2: layer-0 embedding of first-hop neighbors, 10 chunks of 2048 rows
  for (int c = 0; c < 10; c++)
    run_attn(0, neighbors1 + c * NB, nullptr, edge_times1 + c * NB,
             neighbors2 + (size_t)c * NB * KNB, nullptr,
             edge_idxs2 + (size_t)c * NB * KNB, edge_times2 + (size_t)c * NB * KNB,
             nbr1 + (size_t)c * NB * FDIM);
  // call 3: layer-1 embedding (direct src/neighbor features)
  run_attn(1, nullptr, src1, timestamps, neighbors1, nbr1,
           edge_idxs1, edge_times1, emb);

  // community projector
  k_cvt_bf16<<<(NB * FDIM + 255) / 256, 256, 0, stream>>>(emb, embB, NB * FDIM);
  launch_gemm(stream, embB, FDIM, Wp1T, FDIM, hp, FDIM, bp1, NB, FDIM, FDIM, GF_RELU);
  k_final<<<(NB * 32 + 255) / 256, 256, 0, stream>>>(hp, Wp2, bp2, (float*)d_out, NB);
}
